// SingleNonLinear2d_19378892439714
// MI455X (gfx1250) — matrix operand order repaired and checked by reading, not compiled, not run on hardware
//
#include <hip/hip_runtime.h>

typedef __attribute__((ext_vector_type(16))) _Float16 v16h;
typedef __attribute__((ext_vector_type(8)))  float    v8f;

static constexpr int NH = 512, NW = 512, NB = 16;
static constexpr int NHW = NH * NW;
static constexpr int NSTEPS = 8;
static constexpr float FDT = 0.01f;
static constexpr int TILE_X = 64, TILE_Y = 16;
static constexpr int TSX = TILE_X + 4;   // 68
static constexpr int TSY = TILE_Y + 4;   // 20

union AF { v16h v; _Float16 h[16]; };

// ---------------- coe field: [5,H,W] tensor-product quadratic Lagrange ----
__global__ __launch_bounds__(256) void coe_field_kernel(
    const float* __restrict__ coe_w,   // [5,17,17]
    const float* __restrict__ xy,      // [H,W,2]
    float* __restrict__ coe_out)       // [5,H,W]
{
  __shared__ float wl[5 * 17 * 17];
  const int tid = threadIdx.x;
  for (int i = tid; i < 5 * 17 * 17; i += 256) wl[i] = coe_w[i];
  __syncthreads();

  const int p = blockIdx.x * 256 + tid;          // pixel index
  const float TWO_PI = 6.28318530717958647692f;
  const float x0 = xy[2 * p + 0];
  const float x1 = xy[2 * p + 1];

  float s0 = x0 * (8.0f / TWO_PI);
  float c0f = floorf(s0); c0f = fminf(fmaxf(c0f, 0.f), 7.f);
  float t = s0 - c0f; const int a0 = 2 * (int)c0f;
  float LA[3] = { 2.f*(t-0.5f)*(t-1.f), -4.f*t*(t-1.f), 2.f*t*(t-0.5f) };

  float s1 = x1 * (8.0f / TWO_PI);
  float c1f = floorf(s1); c1f = fminf(fmaxf(c1f, 0.f), 7.f);
  float u = s1 - c1f; const int b0 = 2 * (int)c1f;
  float LB[3] = { 2.f*(u-0.5f)*(u-1.f), -4.f*u*(u-1.f), 2.f*u*(u-0.5f) };

  for (int c = 0; c < 5; ++c) {
    const float* wb = wl + c * 289;
    float acc = 0.f;
    #pragma unroll
    for (int i = 0; i < 3; ++i)
      #pragma unroll
      for (int j = 0; j < 3; ++j)
        acc = fmaf(wb[(a0 + i) * 17 + (b0 + j)], LA[i] * LB[j], acc);
    coe_out[c * NHW + p] = acc;
  }
}

// ---------------- fused PDE step: 7-channel 5x5 conv via WMMA + combine ----
__global__ __launch_bounds__(128) void step_kernel(
    const float* __restrict__ uin,   // [B,H,W]
    float* __restrict__ uout,        // [B,H,W]
    const float* __restrict__ coe,   // [5,H,W]
    const float* __restrict__ nlw,   // [161]
    const float* __restrict__ idk,   // [25]
    const float* __restrict__ fdk)   // [6,25]
{
  __shared__ _Float16 tile[TSY * TSX];    // u tile + halo, f16
  __shared__ float scr[4][16 * 17];       // per-wave D spill (pad 17 vs banks)
  __shared__ float nls[161];

  const int tid  = threadIdx.x;
  const int wave = tid >> 5;
  const int lane = tid & 31;
  const int m    = lane & 15;   // A: pixel row M; B/D: column N (channel)
  const int hf   = lane >> 4;

  const int x0 = blockIdx.x * TILE_X;
  const int y0 = blockIdx.y * TILE_Y;
  const int b  = blockIdx.z;

  // Filter B fragment (32x16, K=tap, N=channel). Per 16-bit B layout:
  // lanes 0-15 hold K=0..7,16..23 of column N=lane, lanes 16-31 hold K=8..15,24..31.
  AF bf;
  {
    const float* fp = (m == 0) ? idk : (fdk + (m - 1) * 25);
    #pragma unroll
    for (int e = 0; e < 16; ++e) {
      const int K = (e < 8) ? e + (hf << 3) : e + 8 + (hf << 3);
      float v = 0.f;
      if (K < 25 && m < 7) v = fp[K];
      bf.h[e] = (_Float16)v;
    }
  }

  // Cooperative load of u tile (+2 halo), zero (Dirichlet) padding, f32->f16.
  const float* ub = uin + (size_t)b * NHW;
  for (int i = tid; i < TSY * TSX; i += 128) {
    const int r = i / TSX, c = i - r * TSX;
    const int gy = y0 + r - 2, gx = x0 + c - 2;
    float v = 0.f;
    if (gy >= 0 && gy < NH && gx >= 0 && gx < NW) v = ub[gy * NW + gx];
    tile[i] = (_Float16)v;
  }
  if (tid < 161) nls[tid] = nlw[tid];
  __syncthreads();

  for (int rr = 0; rr < 4; ++rr) {
    const int yl = wave * 4 + rr;              // output row within tile
    for (int g = 0; g < 4; ++g) {
      const int xl = g * 16;
      const int xb = xl + m;                   // pixel column in tile coords

      // A fragment (16x32, M=pixel, K=tap=dy*5+dx). 16-bit A layout:
      // lanes 0-15: VGPR0-3 K=0..7, VGPR4-7 K=16..23
      // lanes 16-31: VGPR0-3 K=8..15, VGPR4-7 K=24..31
      AF a;
      if (hf == 0) {
        #pragma unroll
        for (int e = 0; e < 16; ++e) {
          const int K = (e < 8) ? e : e + 8;           // 0..7, 16..23
          a.h[e] = tile[(yl + K / 5) * TSX + xb + K % 5];
        }
      } else {
        #pragma unroll
        for (int e = 0; e < 16; ++e) {
          const int K = (e < 8) ? e + 8 : e + 16;      // 8..15, 24..31
          a.h[e] = (K < 25) ? tile[(yl + K / 5) * TSX + xb + K % 5]
                            : (_Float16)0.f;
        }
      }

      v8f cz = {};
      v8f d = __builtin_amdgcn_wmma_f32_16x16x32_f16(
          false, a.v, false, bf.v, (short)0, cz, false, false);

      // Spill D (M=hf*8+r, N=m) to padded LDS scratch.
      #pragma unroll
      for (int r = 0; r < 8; ++r)
        scr[wave][(hf * 8 + r) * 17 + m] = d[r];

      // Combine: one lane per pixel (intra-wave LDS is in-order, no barrier).
      if (lane < 16) {
        const float* s = &scr[wave][lane * 17];
        const float uid = s[0];
        const float xv  = s[1];
        const int y = y0 + yl, x = x0 + xl + lane;
        const float* cb = coe + (size_t)y * NW + x;
        float acc =          cb[0 * NHW] * s[2];
        acc = fmaf(cb[1 * NHW], s[3], acc);
        acc = fmaf(cb[2 * NHW], s[4], acc);
        acc = fmaf(cb[3 * NHW], s[5], acc);
        acc = fmaf(cb[4 * NHW], s[6], acc);
        // degree-4 Lagrange interp of nl_w over [-15,15], 40 cells
        const float sN = (xv + 15.f) * (4.f / 3.f);
        float cf = floorf(sN); cf = fminf(fmaxf(cf, 0.f), 39.f);
        const float t  = sN - cf;
        const float t1 = t - 0.25f, t2 = t - 0.5f, t3 = t - 0.75f, t4 = t - 1.f;
        const float L0 = t1 * t2 * t3 * t4 * (256.f / 24.f);
        const float L1 = t  * t2 * t3 * t4 * (-256.f / 6.f);
        const float L2 = t  * t1 * t3 * t4 * (256.f / 4.f);
        const float L3 = t  * t1 * t2 * t4 * (-256.f / 6.f);
        const float L4 = t  * t1 * t2 * t3 * (256.f / 24.f);
        const float* wv = nls + ((int)cf) * 4;
        const float nl = wv[0]*L0 + wv[1]*L1 + wv[2]*L2 + wv[3]*L3 + wv[4]*L4;
        uout[(size_t)b * NHW + (size_t)y * NW + x] = uid + FDT * (acc + nl);
      }
    }
  }
}

extern "C" void kernel_launch(void* const* d_in, const int* in_sizes, int n_in,
                              void* d_out, int out_size, void* d_ws, size_t ws_size,
                              hipStream_t stream)
{
  const float* init = (const float*)d_in[0];
  const float* idk  = (const float*)d_in[1];
  const float* fdk  = (const float*)d_in[2];
  const float* coew = (const float*)d_in[3];
  const float* nlw  = (const float*)d_in[4];
  const float* xy   = (const float*)d_in[5];
  float* out = (float*)d_out;

  float* coe  = (float*)d_ws;                    // 5*NHW f32
  float* ubuf = (float*)d_ws + (size_t)5 * NHW;  // NB*NHW f32 ping buffer

  coe_field_kernel<<<NHW / 256, 256, 0, stream>>>(coew, xy, coe);

  dim3 grid(NW / TILE_X, NH / TILE_Y, NB);
  // Ping-pong so step NSTEPS-1 lands in d_out (NSTEPS is even).
  for (int s = 0; s < NSTEPS; ++s) {
    float* o = (((NSTEPS - 1 - s) & 1) == 0) ? out : ubuf;
    const float* in =
        (s == 0) ? init
                 : ((((NSTEPS - s) & 1) == 0) ? (const float*)out
                                              : (const float*)ubuf);
    step_kernel<<<grid, 128, 0, stream>>>(in, o, coe, nlw, idk, fdk);
  }
  (void)in_sizes; (void)n_in; (void)out_size; (void)ws_size;
}
